// MaxSA_73332271612202
// MI455X (gfx1250) — matrix operand order repaired and checked by reading, not compiled, not run on hardware
//
#include <hip/hip_runtime.h>
#include <hip/hip_bf16.h>
#include <cstdint>
#include <cstddef>

typedef __bf16 v16bf __attribute__((ext_vector_type(16)));
typedef float  v8f   __attribute__((ext_vector_type(8)));

union Frag {
    v16bf bf;
    unsigned short u[16];
    unsigned int   w[8];
};

__device__ __forceinline__ unsigned short f2bf(float f) {
    unsigned int u = __float_as_uint(f);
    u += 0x7FFFu + ((u >> 16) & 1u);   // round-to-nearest-even
    return (unsigned short)(u >> 16);
}

#define SWZ_F(x, imm) __int_as_float(__builtin_amdgcn_ds_swizzle(__float_as_int(x), imm))

__device__ __forceinline__ float redmax16(float x) {   // max over 16-lane groups
    x = fmaxf(x, SWZ_F(x, 0x041f));
    x = fmaxf(x, SWZ_F(x, 0x081f));
    x = fmaxf(x, SWZ_F(x, 0x101f));
    x = fmaxf(x, SWZ_F(x, 0x201f));
    return x;
}
__device__ __forceinline__ float redsum16(float x) {   // sum over 16-lane groups
    x += SWZ_F(x, 0x041f);
    x += SWZ_F(x, 0x081f);
    x += SWZ_F(x, 0x101f);
    x += SWZ_F(x, 0x201f);
    return x;
}
__device__ __forceinline__ float redsum32(float x) {   // sum over all 32 lanes
    x += SWZ_F(x, 0x041f);
    x += SWZ_F(x, 0x081f);
    x += SWZ_F(x, 0x101f);
    x += SWZ_F(x, 0x201f);
    x += SWZ_F(x, 0x401f);
    return x;
}

// ---------------------------------------------------------------------------
// Weight prep: fp32 [Nout, Kin] row-major -> bf16 packed directly in WMMA
// B-fragment order: uint index = ((kb*(Nout/16)+nb)*32 + lane)*8 + v,
// holding elements (k = kb*32 + (lane>>4)*16 + 2v {+1}, n = nb*16 + (lane&15)).
// ---------------------------------------------------------------------------
__global__ void wprep(const float* __restrict__ w, unsigned int* __restrict__ wt,
                      int Nout, int Kin) {
    int i = blockIdx.x * blockDim.x + threadIdx.x;
    if (i >= (Nout * Kin) / 2) return;
    int v    = i & 7;
    int lane = (i >> 3) & 31;
    int tile = i >> 8;
    int nT16 = Nout >> 4;
    int nb = tile % nT16, kb = tile / nT16;
    int n = nb * 16 + (lane & 15);
    int k = kb * 32 + (lane >> 4) * 16 + 2 * v;
    unsigned int lo = f2bf(w[(size_t)n * Kin + k]);
    unsigned int hh = f2bf(w[(size_t)n * Kin + k + 1]);
    wt[i] = lo | (hh << 16);
}

// ---------------------------------------------------------------------------
// Tiled bf16 WMMA GEMM: C[M,N] = A[M,K] * Bfrag (+bias[n]) (+resid[m,n])
// block = 256 threads (8 waves). Tile: 128(M) x 128(N), K-step 32.
// All 8 B fragments are fetched into distinct registers first (one load
// clause), then the 8-WMMA chain runs, overlapping VMEM latency with math.
// ---------------------------------------------------------------------------
template <bool A_BF16, bool OUT_BF16>
__global__ __launch_bounds__(256) void gemm_wmma(
    const void* __restrict__ Ap, const unsigned int* __restrict__ Bfrag,
    const float* __restrict__ bias, const float* __restrict__ resid,
    void* __restrict__ Cp, int M, int N, int K) {
    __shared__ unsigned short As[128][34];

    const int tid  = threadIdx.x;
    const int lane = tid & 31, wave = tid >> 5;
    const int hi = lane >> 4, ln = lane & 15;
    const int m0 = blockIdx.y * 128;
    const int n0 = blockIdx.x * 128;
    const int nT16 = N >> 4;

    const float*          Af = (const float*)Ap;
    const unsigned short* Ab = (const unsigned short*)Ap;

    v8f acc[8] = {};

    for (int k0 = 0; k0 < K; k0 += 32) {
#pragma unroll
        for (int i = 0; i < 8; ++i) {
            int idx = tid + i * 256;
            int r = idx >> 4, cp = (idx & 15) * 2;
            unsigned int pk;
            if constexpr (A_BF16) {
                pk = *(const unsigned int*)&Ab[(size_t)(m0 + r) * K + k0 + cp];
            } else {
                const float* s = &Af[(size_t)(m0 + r) * K + k0 + cp];
                pk = (unsigned int)f2bf(s[0]) | ((unsigned int)f2bf(s[1]) << 16);
            }
            *(unsigned int*)&As[r][cp] = pk;
        }
        __syncthreads();

        // prefetch all 8 B fragments (16 x b128, one clause)
        const uint4* bq = (const uint4*)(Bfrag +
            ((size_t)(k0 >> 5) * nT16 + (n0 >> 4)) * 256 + lane * 8);
        Frag b[8];
#pragma unroll
        for (int nt = 0; nt < 8; ++nt) {
            uint4 blo = bq[nt * 64];
            uint4 bhi = bq[nt * 64 + 1];
            b[nt].w[0] = blo.x; b[nt].w[1] = blo.y; b[nt].w[2] = blo.z; b[nt].w[3] = blo.w;
            b[nt].w[4] = bhi.x; b[nt].w[5] = bhi.y; b[nt].w[6] = bhi.z; b[nt].w[7] = bhi.w;
        }

        Frag a;
        {
            int m = wave * 16 + ln;
#pragma unroll
            for (int j = 0; j < 8; ++j) {
                int k = hi * 16 + 2 * j;
                a.w[j] = *(const unsigned int*)&As[m][k];
            }
        }

#pragma unroll
        for (int nt = 0; nt < 8; ++nt) {
            acc[nt] = __builtin_amdgcn_wmma_f32_16x16x32_bf16(
                false, a.bf, false, b[nt].bf, (short)0, acc[nt], false, false);
        }
        __syncthreads();
    }

#pragma unroll
    for (int nt = 0; nt < 8; ++nt) {
#pragma unroll
        for (int r = 0; r < 8; ++r) {
            int m = m0 + wave * 16 + r + 8 * hi;
            int n = n0 + nt * 16 + ln;
            float v = acc[nt][r];
            if (bias)  v += bias[n];
            if (resid) v += resid[(size_t)m * N + n];
            if constexpr (OUT_BF16)
                ((unsigned short*)Cp)[(size_t)m * N + n] = f2bf(v);
            else
                ((float*)Cp)[(size_t)m * N + n] = v;
        }
    }
}

// ---------------------------------------------------------------------------
// Flash attention. Block = 128 threads = 4 waves, each wave owns one 16-row
// query tile of the same (window, head); K/V chunks staged in LDS once per
// block. All masking is branchless (clamped indices + cndmask selects);
// for N % 32 == 0 (stage 2) all masking folds away at compile time.
// ---------------------------------------------------------------------------
template <int N, int WSZ>
__global__ __launch_bounds__(128) void attn_wmma(
    const unsigned short* __restrict__ qkv, const float* __restrict__ rel_table,
    unsigned short* __restrict__ out) {
    constexpr int  SPAN = 2 * WSZ - 1;
    constexpr bool FULL = (N % 32 == 0);
    __shared__ unsigned short Ks[32][32];
    __shared__ unsigned short Vs[32][32];
    __shared__ unsigned short P[4][16][32];

    const int tid  = threadIdx.x;
    const int lane = tid & 31, wave = tid >> 5;
    const int hi = lane >> 4, ln = lane & 15;
    const int qt = blockIdx.x * 4 + wave;
    const int head = blockIdx.y, win = blockIdx.z;
    const size_t base = (size_t)win * N * 384 + head * 32;
    const float scale = 5.65685424949238f;  // sqrt(head_dim), per reference

    Frag qf;
    {
        int qm = qt * 16 + ln;
        qm = (qm > N - 1) ? (N - 1) : qm;
#pragma unroll
        for (int j = 0; j < 8; ++j) {
            int k = hi * 16 + 2 * j;
            qf.w[j] = *(const unsigned int*)(qkv + base + (size_t)qm * 384 + k);
        }
    }

    float mrow[8], lrow[8];
    v8f o0 = {}, o1 = {};
#pragma unroll
    for (int r = 0; r < 8; ++r) { mrow[r] = -1e30f; lrow[r] = 0.0f; }

    for (int kc = 0; kc < N; kc += 32) {
        {   // cooperative stage: 128 threads move 32 rows x 64B of K and V
            int rr = tid >> 2, qq = tid & 3;
            int j = kc + rr;
            j = (j > N - 1) ? (N - 1) : j;
            const uint4* kp = (const uint4*)(qkv + base + (size_t)j * 384 + 128);
            const uint4* vp = (const uint4*)(qkv + base + (size_t)j * 384 + 256);
            ((uint4*)Ks[rr])[qq] = kp[qq];
            ((uint4*)Vs[rr])[qq] = vp[qq];
        }
        __syncthreads();

        v8f s0v = {}, s1v = {};
        {
            Frag kf;
#pragma unroll
            for (int v = 0; v < 8; ++v)
                kf.w[v] = *(const unsigned int*)&Ks[ln][hi * 16 + 2 * v];
            s0v = __builtin_amdgcn_wmma_f32_16x16x32_bf16(
                false, qf.bf, false, kf.bf, (short)0, s0v, false, false);
#pragma unroll
            for (int v = 0; v < 8; ++v)
                kf.w[v] = *(const unsigned int*)&Ks[16 + ln][hi * 16 + 2 * v];
            s1v = __builtin_amdgcn_wmma_f32_16x16x32_bf16(
                false, qf.bf, false, kf.bf, (short)0, s1v, false, false);
        }

        const int n0g = kc + ln, n1g = kc + 16 + ln;
        const int n0c = FULL ? n0g : ((n0g > N - 1) ? (N - 1) : n0g);
        const int n1c = FULL ? n1g : ((n1g > N - 1) ? (N - 1) : n1g);
        const int r2a = n0c / WSZ, c2a = n0c % WSZ;
        const int r2b = n1c / WSZ, c2b = n1c % WSZ;
#pragma unroll
        for (int r = 0; r < 8; ++r) {
            int m = qt * 16 + r + 8 * hi;
            int mc = FULL ? m : ((m > N - 1) ? (N - 1) : m);
            int r1 = mc / WSZ, c1 = mc % WSZ;
            float b0 = rel_table[((r1 - r2a + WSZ - 1) * SPAN + (c1 - c2a + WSZ - 1)) * 4 + head];
            float b1 = rel_table[((r1 - r2b + WSZ - 1) * SPAN + (c1 - c2b + WSZ - 1)) * 4 + head];
            float s0 = s0v[r] * scale + b0;
            float s1 = s1v[r] * scale + b1;
            if constexpr (!FULL) {
                s0 = (m < N && n0g < N) ? s0 : -1e30f;
                s1 = (m < N && n1g < N) ? s1 : -1e30f;
            }
            float cm = redmax16(fmaxf(s0, s1));
            float nm = fmaxf(mrow[r], cm);
            float alpha = __expf(mrow[r] - nm);
            float p0 = __expf(s0 - nm);
            float p1 = __expf(s1 - nm);
            float cs = redsum16(p0 + p1);
            lrow[r] = lrow[r] * alpha + cs;
            mrow[r] = nm;
            o0[r] *= alpha;
            o1[r] *= alpha;
            P[wave][r + 8 * hi][ln]      = f2bf(p0);
            P[wave][r + 8 * hi][16 + ln] = f2bf(p1);
        }
        __syncthreads();

        Frag pf;
#pragma unroll
        for (int j = 0; j < 8; ++j) {
            int k = hi * 16 + 2 * j;
            pf.w[j] = *(const unsigned int*)&P[wave][ln][k];
        }
        Frag vf;
#pragma unroll
        for (int v = 0; v < 8; ++v) {
            int kr = hi * 16 + 2 * v;
            vf.u[2 * v]     = Vs[kr][ln];
            vf.u[2 * v + 1] = Vs[kr + 1][ln];
        }
        o0 = __builtin_amdgcn_wmma_f32_16x16x32_bf16(
            false, pf.bf, false, vf.bf, (short)0, o0, false, false);
#pragma unroll
        for (int v = 0; v < 8; ++v) {
            int kr = hi * 16 + 2 * v;
            vf.u[2 * v]     = Vs[kr][16 + ln];
            vf.u[2 * v + 1] = Vs[kr + 1][16 + ln];
        }
        o1 = __builtin_amdgcn_wmma_f32_16x16x32_bf16(
            false, pf.bf, false, vf.bf, (short)0, o1, false, false);
        __syncthreads();
    }

#pragma unroll
    for (int r = 0; r < 8; ++r) {
        int m = qt * 16 + r + 8 * hi;
        float inv = 1.0f / lrow[r];
        if (FULL || m < N) {
            size_t ob = ((size_t)win * N + m) * 128 + head * 32;
            out[ob + ln]      = f2bf(o0[r] * inv);
            out[ob + 16 + ln] = f2bf(o1[r] * inv);
        }
    }
}

// ---------------------------------------------------------------------------
// LayerNorm over C=128, in place. One wave per token, float4 I/O.
// ---------------------------------------------------------------------------
__global__ __launch_bounds__(256) void ln_kernel(float* __restrict__ Y,
                                                 const float* __restrict__ g,
                                                 const float* __restrict__ b) {
    int lane = threadIdx.x & 31;
    int t = blockIdx.x * 8 + (threadIdx.x >> 5);
    float4* row = (float4*)(Y + (size_t)t * 128);
    float4 v = row[lane];
    float s = redsum32(v.x + v.y + v.z + v.w);
    float mu = s * (1.0f / 128.0f);
    float dx = v.x - mu, dy = v.y - mu, dz = v.z - mu, dw = v.w - mu;
    float q = redsum32(dx * dx + dy * dy + dz * dz + dw * dw);
    float rs = rsqrtf(q * (1.0f / 128.0f) + 1e-5f);
    float4 gg = ((const float4*)g)[lane];
    float4 bb = ((const float4*)b)[lane];
    float4 o;
    o.x = dx * rs * gg.x + bb.x;
    o.y = dy * rs * gg.y + bb.y;
    o.z = dz * rs * gg.z + bb.z;
    o.w = dw * rs * gg.w + bb.w;
    row[lane] = o;
}

// ---------------------------------------------------------------------------
// Permutations (faithful to the reference's reshape/transpose quirks)
// ---------------------------------------------------------------------------
__global__ void gather_win7(const float* __restrict__ x, float* __restrict__ X1) {
    int d = blockIdx.x * blockDim.x + threadIdx.x;
    if (d >= 50176 * 128) return;
    int c = d & 127;
    int t = d >> 7;
    int n = t % 49, win = t / 49;
    int p = n / 7, q = n % 7;
    int a1 = win >> 5, a2 = win & 31;
    X1[d] = x[(size_t)c * 50176 + (a1 * 7 + p) * 224 + (a2 * 7 + q)];
}

// stage1 out [1024,49,128] -> stage2 in [49,1024,128] via unbind_win(7)+window_partition(32)
__global__ void perm_stage2(const float* __restrict__ Y1, float* __restrict__ X2) {
    int d = blockIdx.x * blockDim.x + threadIdx.x;
    if (d >= 50176 * 128) return;
    int c2 = d & 127;
    int t2 = d >> 7;
    int tok = t2 & 1023, gw = t2 >> 10;
    int u = tok >> 5, v = tok & 31;
    int gi = gw / 7, gj = gw % 7;
    int h = gi * 32 + u, w = gj * 32 + v;
    int f = h * 28672 + w * 128 + c2;        // flat of [H,W,C] reinterpretation
    int c = f / 50176;
    int r = f % 50176;
    int a1 = r / 1568;  r %= 1568;           // strides of (C, 32, 7, 7, 32)
    int p  = r / 224;   r %= 224;
    int q  = r / 32;
    int a2 = r & 31;
    X2[d] = Y1[(size_t)((a1 * 32 + a2) * 49 + p * 7 + q) * 128 + c];
}

// stage2 out [49,1024,128] -> final [128,224,224] via unbind_win(32)+transpose
__global__ void scatter_out(const float* __restrict__ Y2, float* __restrict__ out) {
    int o = blockIdx.x * blockDim.x + threadIdx.x;
    if (o >= 128 * 50176) return;
    int c2 = o / 50176;
    int rem = o % 50176;
    int h = rem / 224, w = rem % 224;
    int f = h * 28672 + w * 128 + c2;
    int c = f / 50176;
    int r = f % 50176;
    int a = r / 7168;  r %= 7168;            // strides of (C, 7, 32, 32, 7)
    int u = r / 224;   r %= 224;
    int v = r / 7;
    int b = r % 7;
    out[o] = Y2[(size_t)((a * 7 + b) * 1024 + (u * 32 + v)) * 128 + c];
}

// ---------------------------------------------------------------------------
extern "C" void kernel_launch(void* const* d_in, const int* in_sizes, int n_in,
                              void* d_out, int out_size, void* d_ws, size_t ws_size,
                              hipStream_t stream) {
    const float* x = (const float*)d_in[0];
    struct StageW {
        const float *rel, *qkv_w, *proj_w, *proj_b, *ln_g, *ln_b, *fc1_w, *fc1_b, *fc2_w, *fc2_b;
    } sw[2];
    for (int s = 0; s < 2; ++s) {
        int o = 1 + s * 10;
        sw[s] = { (const float*)d_in[o + 0], (const float*)d_in[o + 1],
                  (const float*)d_in[o + 2], (const float*)d_in[o + 3],
                  (const float*)d_in[o + 4], (const float*)d_in[o + 5],
                  (const float*)d_in[o + 6], (const float*)d_in[o + 7],
                  (const float*)d_in[o + 8], (const float*)d_in[o + 9] };
    }

    const int T = 50176;  // 224*224 tokens, = 392 * 128
    char* ws = (char*)d_ws;
    size_t off = 0;
    auto alloc = [&](size_t bytes) {
        void* p = ws + off;
        off = (off + bytes + 255) & ~(size_t)255;
        return p;
    };
    float*          Xbuf = (float*)alloc((size_t)T * 128 * 4);
    unsigned short* QKV  = (unsigned short*)alloc((size_t)T * 384 * 2);
    unsigned short* ATT  = (unsigned short*)alloc((size_t)T * 128 * 2);
    float*          Ybuf = (float*)alloc((size_t)T * 128 * 4);
    unsigned short* FC1  = (unsigned short*)alloc((size_t)T * 512 * 2);
    float*          OUT  = (float*)alloc((size_t)T * 128 * 4);
    unsigned int *Wq[2], *Wp[2], *W1[2], *W2[2];
    for (int s = 0; s < 2; ++s) {
        Wq[s] = (unsigned int*)alloc(128 * 384 * 2);
        Wp[s] = (unsigned int*)alloc(128 * 128 * 2);
        W1[s] = (unsigned int*)alloc(128 * 512 * 2);
        W2[s] = (unsigned int*)alloc(512 * 128 * 2);
    }

    for (int s = 0; s < 2; ++s) {
        wprep<<<(384 * 128 / 2 + 255) / 256, 256, 0, stream>>>(sw[s].qkv_w,  Wq[s], 384, 128);
        wprep<<<(128 * 128 / 2 + 255) / 256, 256, 0, stream>>>(sw[s].proj_w, Wp[s], 128, 128);
        wprep<<<(512 * 128 / 2 + 255) / 256, 256, 0, stream>>>(sw[s].fc1_w,  W1[s], 512, 128);
        wprep<<<(128 * 512 / 2 + 255) / 256, 256, 0, stream>>>(sw[s].fc2_w,  W2[s], 128, 512);
    }

    const int tot = T * 128;
    gather_win7<<<(tot + 255) / 256, 256, 0, stream>>>(x, Xbuf);

    for (int s = 0; s < 2; ++s) {
        // QKV = X @ qkv_w.T  (no bias in reference)
        gemm_wmma<false, true><<<dim3(384 / 128, T / 128), 256, 0, stream>>>(
            Xbuf, Wq[s], nullptr, nullptr, QKV, T, 384, 128);
        // attention (scale * QK^T + rel bias, softmax, @V)
        if (s == 0)
            attn_wmma<49, 7><<<dim3(1, 4, 1024), 128, 0, stream>>>(QKV, sw[s].rel, ATT);
        else
            attn_wmma<1024, 32><<<dim3(16, 4, 49), 128, 0, stream>>>(QKV, sw[s].rel, ATT);
        // proj + bias + residual(X)
        gemm_wmma<true, false><<<dim3(1, T / 128), 256, 0, stream>>>(
            ATT, Wp[s], sw[s].proj_b, Xbuf, Ybuf, T, 128, 128);
        // LayerNorm (post-norm) in place -> Ybuf is both MLP input and shortcut
        ln_kernel<<<T / 8, 256, 0, stream>>>(Ybuf, sw[s].ln_g, sw[s].ln_b);
        // fc1 + bias (no activation in reference)
        gemm_wmma<false, true><<<dim3(512 / 128, T / 128), 256, 0, stream>>>(
            Ybuf, W1[s], sw[s].fc1_b, nullptr, FC1, T, 512, 128);
        // fc2 + bias + residual(LN)
        gemm_wmma<true, false><<<dim3(1, T / 128), 256, 0, stream>>>(
            FC1, W2[s], sw[s].fc2_b, Ybuf, OUT, T, 128, 512);

        if (s == 0)
            perm_stage2<<<(tot + 255) / 256, 256, 0, stream>>>(OUT, Xbuf);
    }

    scatter_out<<<(tot + 255) / 256, 256, 0, stream>>>(OUT, (float*)d_out);
}
